// Net_14740327759943
// MI455X (gfx1250) — matrix operand order repaired and checked by reading, not compiled, not run on hardware
//
#include <hip/hip_runtime.h>
#include <cmath>

#define WAVES 4

typedef __attribute__((ext_vector_type(16))) _Float16 v16h;
typedef __attribute__((ext_vector_type(8)))  _Float16 h8;
typedef __attribute__((ext_vector_type(8)))  float    v8f;

// Catmull-Rom (Keys a=-0.5) weights for exact scale-1/2 (antialiased, 8 taps)
__constant__ float DW8[8] = {
  -0.01171875f, -0.03515625f, 0.11328125f, 0.43359375f,
   0.43359375f,  0.11328125f, -0.03515625f, -0.01171875f};
// and exact scale-2 upsample (4 taps, per output parity)
__constant__ float UW4[2][4] = {
  {-0.0234375f, 0.2265625f, 0.8671875f, -0.0703125f},
  {-0.0703125f, 0.8671875f, 0.2265625f, -0.0234375f}};

__device__ __forceinline__ v16h joinh(h8 lo, h8 hi) {
  return __builtin_shufflevector(lo, hi, 0,1,2,3,4,5,6,7,8,9,10,11,12,13,14,15);
}

__device__ __forceinline__ float bsamp(const float* __restrict__ img, int y, int x,
                                       int H, int W) {
  bool ok = (y >= 0) && (y < H) && (x >= 0) && (x < W);
  int yc = y < 0 ? 0 : (y > H - 1 ? H - 1 : y);
  int xc = x < 0 ? 0 : (x > W - 1 ? W - 1 : x);
  float v = img[yc * W + xc];
  return ok ? v : 0.0f;
}

// ---------------------------------------------------------------------------
// Weight pack: OIHW fp32 -> f16 [ocp][Kt] rows (K = ic*KS*KS + ky*KS + kx),
// zero padded so WMMA A-fragment loads are two 16B vectors per K-chunk.
// ---------------------------------------------------------------------------
__global__ void pack_w_kernel(const float* __restrict__ w, _Float16* __restrict__ wp,
                              int OC, int Ktot, int Kt, long total) {
  long i = (long)blockIdx.x * blockDim.x + threadIdx.x;
  if (i >= total) return;
  int k = (int)(i % Kt);
  int o = (int)(i / Kt);
  float v = (o < OC && k < Ktot) ? w[(long)o * Ktot + k] : 0.0f;
  wp[i] = (_Float16)v;
}

// ---------------------------------------------------------------------------
// Implicit-GEMM conv via V_WMMA_F32_16X16X32_F16.
// Wave = 16 outchannels x 16 pixels. Supports 2-tensor channel concat input.
// ---------------------------------------------------------------------------
template <int KS, int PAD>
__global__ void __launch_bounds__(32 * WAVES)
conv_wmma_kernel(const float* __restrict__ in0, int ch0, long bs0,
                 const float* __restrict__ in1, long bs1,
                 const _Float16* __restrict__ Wp, int Kt, int Ktot,
                 const float* __restrict__ bias, int OC,
                 float* __restrict__ out, long obs,
                 int H, int W, int relu) {
  const int HWl  = H * W;
  const int lane = threadIdx.x & 31;
  const int wave = threadIdx.x >> 5;
  const int ptile = blockIdx.x * WAVES + wave;
  const int col  = lane & 15;
  const int half = lane >> 4;
  const int b    = blockIdx.z;
  const int pix  = ptile * 16 + col;
  const int py   = pix / W;
  const int px   = pix - py * W;
  const float* inb0 = in0 + (long)b * bs0;
  const float* inb1 = in1 ? (in1 + (long)b * bs1) : (const float*)0;
  const _Float16* wrow = Wp + (long)(blockIdx.y * 16 + col) * Kt;

  v8f acc = {};
  for (int kc = 0; kc < Kt; kc += 32) {
    // A fragment: M = col, lane-half selects K 0-15 / 16-31 of chunk
    h8 alo = *(const h8*)(wrow + kc + half * 16);
    h8 ahi = *(const h8*)(wrow + kc + half * 16 + 8);
    v16h af = joinh(alo, ahi);
    __builtin_prefetch(wrow + kc + 32, 0, 0);  // global_prefetch_b8 on next chunk

    // B fragment: N = col, lane-half selects K 0-15 / 16-31 of chunk
    v16h bf;
#pragma unroll
    for (int e = 0; e < 16; ++e) {
      int k = kc + half * 16 + e;
      float v = 0.0f;
      if (k < Ktot) {
        int ic = k / (KS * KS);
        int r  = k - ic * (KS * KS);
        int ky = r / KS;
        int kx = r - ky * KS;
        int y = py + ky - PAD;
        int x = px + kx - PAD;
        if ((unsigned)y < (unsigned)H && (unsigned)x < (unsigned)W) {
          const float* s = (ic < ch0) ? (inb0 + (long)ic * HWl)
                                      : (inb1 + (long)(ic - ch0) * HWl);
          v = s[y * W + x];
        }
      }
      bf[e] = (_Float16)v;
    }
    acc = __builtin_amdgcn_wmma_f32_16x16x32_f16(false, af, false, bf,
                                                 (short)0, acc, false, false);
  }

#pragma unroll
  for (int r = 0; r < 8; ++r) {
    int oc = blockIdx.y * 16 + half * 8 + r;  // C/D layout: M = r + 8*half
    if (oc < OC) {
      float v = acc[r] + bias[oc];
      if (relu) v = (v > 0.0f) ? v : 0.01f * v;
      out[(long)b * obs + (long)oc * HWl + pix] = v;
    }
  }
}

// ---------------------------------------------------------------------------
// Deformable conv 3x3 (64ch in, 64ch out) as WMMA GEMM with K = 64*9 = 576.
// Per-wave: stage 16 pixels x 9 taps (y, x, sigmoid(mask)) in LDS, then
// bilinear-sample B fragments; all 4 OC tiles share the sampled data.
// ---------------------------------------------------------------------------
__global__ void __launch_bounds__(32 * WAVES)
dcn_wmma_kernel(const float* __restrict__ x, long xbs,
                const float* __restrict__ om, long ombs,
                const _Float16* __restrict__ Wp,
                const float* __restrict__ bias,
                float* __restrict__ out, long obs,
                int H, int W) {
  const int HWl = H * W;
  const int Kt  = 576;  // 18 chunks of 32
  __shared__ float taps[WAVES][9][16][3];

  const int lane = threadIdx.x & 31;
  const int wave = threadIdx.x >> 5;
  const int ptile = blockIdx.x * WAVES + wave;
  const int col  = lane & 15;
  const int half = lane >> 4;
  const int b    = blockIdx.z;
  const float* omb = om + (long)b * ombs;
  const float* xb  = x + (long)b * xbs;

  for (int e = lane; e < 144; e += 32) {
    int t  = e / 16;
    int pp = e - t * 16;
    int pix = ptile * 16 + pp;
    int py = pix / W;
    int px = pix - py * W;
    float dy = omb[(long)(2 * t) * HWl + pix];
    float dx = omb[(long)(2 * t + 1) * HWl + pix];
    float mr = omb[(long)(18 + t) * HWl + pix];
    taps[wave][t][pp][0] = (float)py + (float)(t / 3 - 1) + dy;
    taps[wave][t][pp][1] = (float)px + (float)(t % 3 - 1) + dx;
    taps[wave][t][pp][2] = 1.0f / (1.0f + expf(-mr));
  }
  __syncthreads();

  const int pix = ptile * 16 + col;
  v8f acc[4] = {{}, {}, {}, {}};

  for (int kc = 0; kc < Kt; kc += 32) {
    v16h bf;
#pragma unroll
    for (int e = 0; e < 16; ++e) {
      int k = kc + half * 16 + e;
      int c = k / 9;
      int t = k - c * 9;
      float ys = taps[wave][t][col][0];
      float xs = taps[wave][t][col][1];
      float m  = taps[wave][t][col][2];
      float y0f = floorf(ys), x0f = floorf(xs);
      float wy = ys - y0f, wx = xs - x0f;
      int y0 = (int)y0f, x0 = (int)x0f;
      const float* xc = xb + (long)c * HWl;
      float v00 = bsamp(xc, y0,     x0,     H, W);
      float v01 = bsamp(xc, y0,     x0 + 1, H, W);
      float v10 = bsamp(xc, y0 + 1, x0,     H, W);
      float v11 = bsamp(xc, y0 + 1, x0 + 1, H, W);
      float v = (v00 * (1.0f - wy) * (1.0f - wx) + v01 * (1.0f - wy) * wx +
                 v10 * wy * (1.0f - wx)          + v11 * wy * wx) * m;
      bf[e] = (_Float16)v;
    }
#pragma unroll
    for (int oct = 0; oct < 4; ++oct) {
      const _Float16* wrow = Wp + (long)(oct * 16 + col) * Kt + kc;
      h8 alo = *(const h8*)(wrow + half * 16);
      h8 ahi = *(const h8*)(wrow + half * 16 + 8);
      v16h af = joinh(alo, ahi);
      acc[oct] = __builtin_amdgcn_wmma_f32_16x16x32_f16(false, af, false, bf,
                                                        (short)0, acc[oct],
                                                        false, false);
    }
  }

#pragma unroll
  for (int oct = 0; oct < 4; ++oct) {
#pragma unroll
    for (int r = 0; r < 8; ++r) {
      int oc = oct * 16 + half * 8 + r;
      out[(long)b * obs + (long)oc * HWl + pix] = acc[oct][r] + bias[oc];
    }
  }
}

// ---------------------------------------------------------------------------
// Bicubic resize (exact jax.image.resize grids, edge-renormalized per dim)
// ---------------------------------------------------------------------------
__global__ void down_bicubic_kernel(const float* __restrict__ in,
                                    float* __restrict__ out,
                                    int CH, int H, int W) {
  int Ho = H >> 1, Wo = W >> 1;
  long total = (long)2 * CH * Ho * Wo;
  long i = (long)blockIdx.x * blockDim.x + threadIdx.x;
  if (i >= total) return;
  int ox = (int)(i % Wo); long t = i / Wo;
  int oy = (int)(t % Ho); t /= Ho;
  int c  = (int)(t % CH);
  int b  = (int)(t / CH);
  const float* src = in + ((long)b * CH + c) * H * W;
  float acc = 0.0f, wys = 0.0f, wxs = 0.0f;
#pragma unroll
  for (int tx = 0; tx < 8; ++tx) {
    int ix = 2 * ox + tx - 3;
    if ((unsigned)ix < (unsigned)W) wxs += DW8[tx];
  }
#pragma unroll
  for (int ty = 0; ty < 8; ++ty) {
    int iy = 2 * oy + ty - 3;
    if ((unsigned)iy >= (unsigned)H) continue;
    float wy = DW8[ty];
    wys += wy;
    float row = 0.0f;
#pragma unroll
    for (int tx = 0; tx < 8; ++tx) {
      int ix = 2 * ox + tx - 3;
      if ((unsigned)ix < (unsigned)W) row += DW8[tx] * src[iy * W + ix];
    }
    acc += wy * row;
  }
  out[i] = acc / (wys * wxs);
}

__global__ void up_bicubic_kernel(const float* __restrict__ in,
                                  float* __restrict__ out,
                                  int CH, int H, int W) {
  int Ho = H << 1, Wo = W << 1;
  long total = (long)2 * CH * Ho * Wo;
  long i = (long)blockIdx.x * blockDim.x + threadIdx.x;
  if (i >= total) return;
  int ox = (int)(i % Wo); long t = i / Wo;
  int oy = (int)(t % Ho); t /= Ho;
  int c  = (int)(t % CH);
  int b  = (int)(t / CH);
  const float* src = in + ((long)b * CH + c) * H * W;
  int jy = oy >> 1, py = oy & 1;
  int jx = ox >> 1, qx = ox & 1;
  float acc = 0.0f, wys = 0.0f, wxs = 0.0f;
#pragma unroll
  for (int tx = 0; tx < 4; ++tx) {
    int ix = jx - 2 + qx + tx;
    if ((unsigned)ix < (unsigned)W) wxs += UW4[qx][tx];
  }
#pragma unroll
  for (int ty = 0; ty < 4; ++ty) {
    int iy = jy - 2 + py + ty;
    if ((unsigned)iy >= (unsigned)H) continue;
    float wy = UW4[py][ty];
    wys += wy;
    float row = 0.0f;
#pragma unroll
    for (int tx = 0; tx < 4; ++tx) {
      int ix = jx - 2 + qx + tx;
      if ((unsigned)ix < (unsigned)W) row += UW4[qx][tx] * src[iy * W + ix];
    }
    acc += wy * row;
  }
  out[i] = acc / (wys * wxs);
}

// ---------------------------------------------------------------------------
// Temporal attention: m = sigmoid(<ne, ce>_c); TAM[:,f]=m; TAS[:,f*64:]=ac*m
// ---------------------------------------------------------------------------
__global__ void attn_kernel(const float* __restrict__ ac, const float* __restrict__ ne,
                            const float* __restrict__ ce, float* __restrict__ tam,
                            float* __restrict__ tas, int f, int HW) {
  long i = (long)blockIdx.x * blockDim.x + threadIdx.x;
  if (i >= 2L * HW) return;
  int b = (int)(i / HW);
  int pix = (int)(i % HW);
  const float* neb = ne + (long)b * 64 * HW + pix;
  const float* ceb = ce + (long)b * 64 * HW + pix;
  float s = 0.0f;
  for (int c = 0; c < 64; ++c) s += neb[(long)c * HW] * ceb[(long)c * HW];
  float m = 1.0f / (1.0f + expf(-s));
  tam[((long)b * 5 + f) * HW + pix] = m;
  const float* acb = ac + (long)b * 64 * HW + pix;
  float* tb = tas + ((long)b * 320 + f * 64) * HW + pix;
  for (int c = 0; c < 64; ++c) tb[(long)c * HW] = acb[(long)c * HW] * m;
}

// ---------------------------------------------------------------------------
// Host orchestration
// ---------------------------------------------------------------------------
static inline int rup(int x, int m) { return (x + m - 1) / m * m; }

extern "C" void kernel_launch(void* const* d_in, const int* in_sizes, int n_in,
                              void* d_out, int out_size, void* d_ws, size_t ws_size,
                              hipStream_t stream) {
  (void)in_sizes; (void)out_size; (void)ws_size;
  if (n_in < 41) return;
  const int B = 2, C = 64, H = 128, W = 128;
  const int HW = H * W;
  const int H2 = 64, W2 = 64, HW2 = H2 * W2;
  const int H4 = 32, W4 = 32, HW4 = H4 * W4;

  const float* feat[5];
  for (int i = 0; i < 5; ++i) feat[i] = (const float*)d_in[i];
  const float* P[41];
  for (int i = 0; i < 41; ++i) P[i] = (const float*)d_in[i];
  const float* omWt[4] = {P[5], P[7], P[9], P[11]};   // omL1, omL2, omL3, omC
  const float* omBs[4] = {P[6], P[8], P[10], P[12]};
  const float *f23W = P[13], *f23B = P[14], *f12W = P[15], *f12B = P[16];
  const float *a23W = P[17], *a23B = P[18], *a12W = P[19], *a12B = P[20];
  const float* dcWt[4] = {P[21], P[23], P[25], P[27]};  // dcn1, dcn2, dcn3, dcnC
  const float* dcBs[4] = {P[22], P[24], P[26], P[28]};
  const float *te1W = P[29], *te1B = P[30], *te2W = P[31], *te2B = P[32];
  const float *fc1W = P[33], *fc1B = P[34], *fc2W = P[35], *fc2B = P[36];
  const float *st1W = P[37], *st1B = P[38], *st2W = P[39], *st2B = P[40];

  char* ws = (char*)d_ws;
  size_t off = 0;
  auto allocF = [&](size_t n) -> float* {
    float* p = (float*)(ws + off);
    off += ((n * 4 + 255) / 256) * 256;
    return p;
  };
  auto allocH = [&](size_t n) -> _Float16* {
    _Float16* p = (_Float16*)(ws + off);
    off += ((n * 2 + 255) / 256) * 256;
    return p;
  };

  struct PW { const _Float16* p; int Kt, Ktot, OC, ocp; };
  auto pack = [&](const float* w, int OC, int Ktot) -> PW {
    int Kt = rup(Ktot, 32), ocp = rup(OC, 16);
    _Float16* wp = allocH((size_t)ocp * Kt);
    long tot = (long)ocp * Kt;
    pack_w_kernel<<<dim3((unsigned)((tot + 255) / 256)), dim3(256), 0, stream>>>(
        w, wp, OC, Ktot, Kt, tot);
    return PW{wp, Kt, Ktot, OC, ocp};
  };

  PW pwOm[4], pwDc[4];
  for (int i = 0; i < 4; ++i) pwOm[i] = pack(omWt[i], 27, 2 * C * 25);
  PW pwF23 = pack(f23W, 27, 54), pwF12 = pack(f12W, 27, 54);
  PW pwA23 = pack(a23W, C, 2 * C), pwA12 = pack(a12W, C, 2 * C);
  for (int i = 0; i < 4; ++i) pwDc[i] = pack(dcWt[i], C, C * 9);
  PW pwT1 = pack(te1W, C, C * 9), pwT2 = pack(te2W, C, C * 9);
  PW pwFc1 = pack(fc1W, C, 5 * C * 9), pwFc2 = pack(fc2W, C, C);
  PW pwS1 = pack(st1W, C / 2, 270 * 9), pwS2 = pack(st2W, C / 2, 5 * 9);

  auto conv = [&](int KS, const float* in0, int ch0, long bs0,
                  const float* in1, long bs1,
                  const PW& pw, const float* bias,
                  float* out, long obs, int h, int w, int relu) {
    dim3 grid((h * w) / (16 * WAVES), pw.ocp / 16, B);
    dim3 block(32 * WAVES);
    if (KS == 5)
      conv_wmma_kernel<5, 2><<<grid, block, 0, stream>>>(
          in0, ch0, bs0, in1, bs1, pw.p, pw.Kt, pw.Ktot, bias, pw.OC, out, obs, h, w, relu);
    else if (KS == 3)
      conv_wmma_kernel<3, 1><<<grid, block, 0, stream>>>(
          in0, ch0, bs0, in1, bs1, pw.p, pw.Kt, pw.Ktot, bias, pw.OC, out, obs, h, w, relu);
    else
      conv_wmma_kernel<1, 0><<<grid, block, 0, stream>>>(
          in0, ch0, bs0, in1, bs1, pw.p, pw.Kt, pw.Ktot, bias, pw.OC, out, obs, h, w, relu);
  };
  auto dcn = [&](const float* x, long xbs, const float* om, long ombs,
                 const PW& pw, const float* bias, float* out, long obs,
                 int h, int w) {
    dim3 grid((h * w) / (16 * WAVES), 1, B);
    dcn_wmma_kernel<<<grid, dim3(32 * WAVES), 0, stream>>>(
        x, xbs, om, ombs, pw.p, bias, out, obs, h, w);
  };
  auto down = [&](const float* in, float* out, int ch, int h, int w) {
    long tot = (long)B * ch * (h / 2) * (w / 2);
    down_bicubic_kernel<<<dim3((unsigned)((tot + 255) / 256)), dim3(256), 0, stream>>>(
        in, out, ch, h, w);
  };
  auto up = [&](const float* in, float* out, int ch, int h, int w) {
    long tot = (long)B * ch * (h * 2) * (w * 2);
    up_bicubic_kernel<<<dim3((unsigned)((tot + 255) / 256)), dim3(256), 0, stream>>>(
        in, out, ch, h, w);
  };

  // ----- workspace tensors -----
  float* C2   = allocF((size_t)B * C * HW2);
  float* C3   = allocF((size_t)B * C * HW4);
  float* N2   = allocF((size_t)B * C * HW2);
  float* N3   = allocF((size_t)B * C * HW4);
  float* OM3  = allocF((size_t)B * 27 * HW4);
  float* OM2R = allocF((size_t)B * 27 * HW2);
  float* UP3  = allocF((size_t)B * 27 * HW2);
  float* OM2  = allocF((size_t)B * 27 * HW2);
  float* OM1R = allocF((size_t)B * 27 * HW);
  float* UP2  = allocF((size_t)B * 27 * HW);
  float* A3   = allocF((size_t)B * C * HW4);
  float* A2R  = allocF((size_t)B * C * HW2);
  float* UPA3 = allocF((size_t)B * C * HW2);
  float* A2   = allocF((size_t)B * C * HW2);
  float* A1R  = allocF((size_t)B * C * HW);
  float* UPA2 = allocF((size_t)B * C * HW);
  float* A1   = allocF((size_t)B * C * HW);
  float* OFFS = allocF((size_t)B * 270 * HW);
  float* ACb  = allocF((size_t)5 * B * C * HW);
  float* CE   = allocF((size_t)B * C * HW);
  float* NE   = allocF((size_t)B * C * HW);
  float* TAM  = allocF((size_t)B * 5 * HW);
  float* TAS  = allocF((size_t)B * 5 * C * HW);
  float* FM1  = allocF((size_t)B * C * HW);

  const long bsC1 = (long)C * HW, bsC2 = (long)C * HW2, bsC4 = (long)C * HW4;
  const long bs27_1 = (long)27 * HW, bs27_2 = (long)27 * HW2, bs27_4 = (long)27 * HW4;
  const long bsOFF = (long)270 * HW;

  // shared pyramid of the center (current) frame
  down(feat[2], C2, C, H, W);
  down(C2, C3, C, H2, W2);

  for (int f = 0; f < 5; ++f) {
    const float* n1 = feat[f];
    down(n1, N2, C, H, W);
    down(N2, N3, C, H2, W2);

    conv(5, C3, C, bsC4, N3, bsC4, pwOm[2], omBs[2], OM3, bs27_4, H4, W4, 0);   // omL3
    conv(5, C2, C, bsC2, N2, bsC2, pwOm[1], omBs[1], OM2R, bs27_2, H2, W2, 0);  // omL2
    up(OM3, UP3, 27, H4, W4);
    conv(1, OM2R, 27, bs27_2, UP3, bs27_2, pwF23, f23B, OM2, bs27_2, H2, W2, 0);
    conv(5, feat[2], C, bsC1, n1, bsC1, pwOm[0], omBs[0], OM1R, bs27_1, H, W, 0);  // omL1
    up(OM2, UP2, 27, H2, W2);
    float* OM1 = OFFS + (size_t)f * 54 * HW;  // om1 -> offs slab, channels [f*54, +27)
    conv(1, OM1R, 27, bs27_1, UP2, bs27_1, pwF12, f12B, OM1, bsOFF, H, W, 0);

    dcn(N3, bsC4, OM3, bs27_4, pwDc[2], dcBs[2], A3, bsC4, H4, W4);
    dcn(N2, bsC2, OM2, bs27_2, pwDc[1], dcBs[1], A2R, bsC2, H2, W2);
    up(A3, UPA3, C, H4, W4);
    conv(1, A2R, C, bsC2, UPA3, bsC2, pwA23, a23B, A2, bsC2, H2, W2, 0);
    dcn(n1, bsC1, OM1, bsOFF, pwDc[0], dcBs[0], A1R, bsC1, H, W);
    up(A2, UPA2, C, H2, W2);
    conv(1, A1R, C, bsC1, UPA2, bsC1, pwA12, a12B, A1, bsC1, H, W, 0);

    float* OMC = OFFS + (size_t)(f * 54 + 27) * HW;  // omc -> offs slab
    conv(5, feat[2], C, bsC1, A1, bsC1, pwOm[3], omBs[3], OMC, bsOFF, H, W, 0);
    float* ACf = ACb + (size_t)f * B * C * HW;
    dcn(A1, bsC1, OMC, bsOFF, pwDc[3], dcBs[3], ACf, bsC1, H, W);
  }

  // temporal attention + fusion
  const float* AC2 = ACb + (size_t)2 * B * C * HW;
  conv(3, AC2, C, bsC1, nullptr, 0, pwT2, te2B, CE, bsC1, H, W, 0);
  for (int f = 0; f < 5; ++f) {
    const float* ACf = ACb + (size_t)f * B * C * HW;
    conv(3, ACf, C, bsC1, nullptr, 0, pwT1, te1B, NE, bsC1, H, W, 0);
    long tot = (long)B * HW;
    attn_kernel<<<dim3((unsigned)((tot + 255) / 256)), dim3(256), 0, stream>>>(
        ACf, NE, CE, TAM, TAS, f, HW);
  }
  conv(3, TAS, 5 * C, (long)5 * C * HW, nullptr, 0, pwFc1, fc1B, FM1, bsC1, H, W, 1);

  float* outFM = (float*)d_out;                         // (B, 64, H, W)
  conv(1, FM1, C, bsC1, nullptr, 0, pwFc2, fc2B, outFM, bsC1, H, W, 1);

  float* outS = outFM + (size_t)B * C * HW;             // (B, 64, H, W): s1|s2
  conv(3, OFFS, 270, bsOFF, nullptr, 0, pwS1, st1B, outS, bsC1, H, W, 1);
  conv(3, TAM, 5, (long)5 * HW, nullptr, 0, pwS2, st2B, outS + (size_t)32 * HW,
       bsC1, H, W, 1);
}
